// TFTMultiHeadAttention_51127290692201
// MI455X (gfx1250) — matrix operand order repaired and checked by reading, not compiled, not run on hardware
//
#include <hip/hip_runtime.h>

#define B_  2
#define T_  2048
#define H_  8
#define DM_ 512
#define DK_ 64

typedef __attribute__((ext_vector_type(16))) __bf16 v16bf;
typedef __attribute__((ext_vector_type(8)))  float  v8f;

union BfFrag {
  v16bf v;
  unsigned short s[16];
};
static_assert(sizeof(BfFrag) == 32, "frag size");

__device__ __forceinline__ unsigned short f32_to_bf16(float f) {
  unsigned int u = __float_as_uint(f);
  u += 0x7FFFu + ((u >> 16) & 1u);   // round-to-nearest-even
  return (unsigned short)(u >> 16);
}

__device__ __forceinline__ v8f wmma_bf16(const BfFrag& A, const BfFrag& Bf, v8f C) {
  // D = A(16x32 bf16) x B(32x16 bf16) + C(16x16 f32)
  return __builtin_amdgcn_wmma_f32_16x16x32_bf16(false, A.v, false, Bf.v,
                                                 (short)0, C, false, false);
}

// A-fragment element e -> K index within the 32-wide chunk (16-bit A 16x32, wave32).
// Lane half r: element e holds K = 16r + e (contiguous order, matches B fills).
__device__ __forceinline__ int a_k(int e, int r) {
  return 16 * r + e;
}

__device__ __forceinline__ float redmax16(float v) {
#pragma unroll
  for (int o = 1; o < 16; o <<= 1) v = fmaxf(v, __shfl_xor(v, o, 32));
  return v;
}
__device__ __forceinline__ float redsum16(float v) {
#pragma unroll
  for (int o = 1; o < 16; o <<= 1) v += __shfl_xor(v, o, 32);
  return v;
}

// A chunk from bf16 row pointer p (row-major, chunk already offset). Contiguous order (k = 16r + e), same as the B fills.
__device__ __forceinline__ void load_a_bf16(BfFrag& A, const unsigned short* p, int r) {
#pragma unroll
  for (int e = 0; e < 8; ++e) A.s[e] = p[16 * r + e];
#pragma unroll
  for (int e = 0; e < 8; ++e) A.s[8 + e] = p[16 * r + 8 + e];
}
// B chunk where K runs contiguously in memory at p: b[e] = p[16*r + e].
__device__ __forceinline__ void load_bT_bf16(BfFrag& Bf, const unsigned short* p, int r) {
#pragma unroll
  for (int e = 0; e < 16; ++e) Bf.s[e] = p[16 * r + e];
}

// ---------------- Kernel 1: projections (q@Wq per head, k@Wk per head, v@Wv shared) --------
__global__ __launch_bounds__(32) void proj_kernel(
    const float* __restrict__ q, const float* __restrict__ k, const float* __restrict__ v,
    const float* __restrict__ Wq, const float* __restrict__ Wk, const float* __restrict__ Wv,
    unsigned short* __restrict__ qs, unsigned short* __restrict__ ks,
    unsigned short* __restrict__ vs) {
  const int tb  = blockIdx.x * 16;
  const int job = blockIdx.y;            // 0..7: Q heads, 8..15: K heads, 16: V shared
  const int b   = blockIdx.z;

  const float* X; const float* W; unsigned short* out;
  if (job < H_) {
    X = q + (size_t)b * T_ * DM_;  W = Wq + (size_t)job * DM_ * DK_;
    out = qs + ((size_t)(b * H_ + job) * T_) * DK_;
  } else if (job < 2 * H_) {
    X = k + (size_t)b * T_ * DM_;  W = Wk + (size_t)(job - H_) * DM_ * DK_;
    out = ks + ((size_t)(b * H_ + job - H_) * T_) * DK_;
  } else {
    X = v + (size_t)b * T_ * DM_;  W = Wv;
    out = vs + (size_t)b * T_ * DK_;
  }

  const int lane = threadIdx.x;
  const int r = lane >> 4, ln = lane & 15;

  v8f z = {0.f, 0.f, 0.f, 0.f, 0.f, 0.f, 0.f, 0.f};
  v8f acc[4];
#pragma unroll
  for (int d = 0; d < 4; ++d) acc[d] = z;

  const float* xrow = X + (size_t)(tb + ln) * DM_;
  for (int kc = 0; kc < DM_ / 32; ++kc) {
    BfFrag A;
    const float* p = xrow + kc * 32;
#pragma unroll
    for (int e = 0; e < 8; ++e) A.s[e] = f32_to_bf16(p[16 * r + e]);
#pragma unroll
    for (int e = 0; e < 8; ++e) A.s[8 + e] = f32_to_bf16(p[16 * r + 8 + e]);
#pragma unroll
    for (int d = 0; d < 4; ++d) {
      BfFrag Bf;
#pragma unroll
      for (int e = 0; e < 16; ++e)
        Bf.s[e] = f32_to_bf16(W[(size_t)(kc * 32 + 16 * r + e) * DK_ + d * 16 + ln]);
      acc[d] = wmma_bf16(A, Bf, acc[d]);
    }
  }
#pragma unroll
  for (int d = 0; d < 4; ++d)
#pragma unroll
    for (int i = 0; i < 8; ++i)
      out[(size_t)(tb + i + 8 * r) * DK_ + d * 16 + ln] = f32_to_bf16(acc[d][i]);
}

// ---------------- Kernel 2: causal attention (softmax fp32, GEMMs bf16 WMMA) ---------------
__global__ __launch_bounds__(32) void attn_kernel(
    const unsigned short* __restrict__ qs, const unsigned short* __restrict__ ks,
    const unsigned short* __restrict__ vs, float* __restrict__ heads,
    float* __restrict__ attn_out) {
  const int qb = blockIdx.x * 16;
  const int h  = blockIdx.y;
  const int b  = blockIdx.z;
  const unsigned short* Q = qs + ((size_t)(b * H_ + h) * T_) * DK_;
  const unsigned short* K = ks + ((size_t)(b * H_ + h) * T_) * DK_;
  const unsigned short* V = vs + (size_t)b * T_ * DK_;
  float* attn = attn_out + ((size_t)(h * B_ + b)) * T_ * T_;   // [H,B,T,T]

  const int lane = threadIdx.x;
  const int r = lane >> 4, ln = lane & 15;
  const float scale = 0.125f;  // 1/sqrt(64)
  v8f z = {0.f, 0.f, 0.f, 0.f, 0.f, 0.f, 0.f, 0.f};

  BfFrag A0, A1;   // Q tile, d_k = 64 -> two 32-wide K chunks
  load_a_bf16(A0, Q + (size_t)(qb + ln) * DK_, r);
  load_a_bf16(A1, Q + (size_t)(qb + ln) * DK_ + 32, r);

  float m_run[8], l_run[8];
#pragma unroll
  for (int i = 0; i < 8; ++i) { m_run[i] = -__builtin_inff(); l_run[i] = 0.f; }

  // ---- Pass 1: streaming online row-max / exp-sum over causal key tiles ----
  for (int kb = 0; kb <= qb; kb += 16) {
    BfFrag B0, B1;
    const unsigned short* krow = K + (size_t)(kb + ln) * DK_;
    load_bT_bf16(B0, krow, r);        // B[kk][n] = K[n][kk] : K runs contiguous in ks row
    load_bT_bf16(B1, krow + 32, r);
    v8f s = z;
    s = wmma_bf16(A0, B0, s);
    s = wmma_bf16(A1, B1, s);
#pragma unroll
    for (int i = 0; i < 8; ++i) {
      const int row = qb + i + 8 * r, col = kb + ln;
      float sv = (col > row) ? -__builtin_inff() : s[i] * scale;
      float tm = redmax16(sv);
      float m_new = fmaxf(m_run[i], tm);
      float pe = __expf(sv - m_new);
      float rs = redsum16(pe);
      l_run[i] = l_run[i] * __expf(m_run[i] - m_new) + rs;
      m_run[i] = m_new;
    }
  }
  float linv[8];
#pragma unroll
  for (int i = 0; i < 8; ++i) linv[i] = 1.f / l_run[i];

  // ---- Pass 2: recompute S, emit attn (fp32) and accumulate O = P x V via WMMA ----
  __shared__ unsigned short Plds[16 * 32];   // P staged bf16 for A-fragment re-layout
  v8f od[4];
#pragma unroll
  for (int d = 0; d < 4; ++d) od[d] = z;

  for (int kb2 = 0; kb2 <= qb; kb2 += 32) {
#pragma unroll
    for (int sub = 0; sub < 2; ++sub) {
      const int kb = kb2 + sub * 16;
      if (kb <= qb) {
        BfFrag B0, B1;
        const unsigned short* krow = K + (size_t)(kb + ln) * DK_;
        load_bT_bf16(B0, krow, r);
        load_bT_bf16(B1, krow + 32, r);
        v8f s = z;
        s = wmma_bf16(A0, B0, s);
        s = wmma_bf16(A1, B1, s);
#pragma unroll
        for (int i = 0; i < 8; ++i) {
          const int row = qb + i + 8 * r, col = kb + ln;
          float pv = (col > row) ? 0.f : __expf(s[i] * scale - m_run[i]) * linv[i];
          attn[(size_t)row * T_ + col] = pv;
          Plds[(i + 8 * r) * 32 + sub * 16 + ln] = f32_to_bf16(pv);
        }
      } else {  // fully-masked second subtile: P must be zero for the PV WMMA
#pragma unroll
        for (int i = 0; i < 8; ++i)
          Plds[(i + 8 * r) * 32 + sub * 16 + ln] = 0;
      }
    }
    __syncthreads();
    BfFrag PA;
#pragma unroll
    for (int e = 0; e < 16; ++e) PA.s[e] = Plds[ln * 32 + a_k(e, r)];
#pragma unroll
    for (int d = 0; d < 4; ++d) {
      BfFrag VB;   // B[kk][n] = V[kb2+kk][d*16+n]
#pragma unroll
      for (int e = 0; e < 16; ++e)
        VB.s[e] = V[(size_t)(kb2 + 16 * r + e) * DK_ + d * 16 + ln];
      od[d] = wmma_bf16(PA, VB, od[d]);
    }
    __syncthreads();
  }

#pragma unroll
  for (int d = 0; d < 4; ++d)
#pragma unroll
    for (int i = 0; i < 8; ++i)
      heads[((size_t)(b * H_ + h) * T_ + qb + i + 8 * r) * DK_ + d * 16 + ln] = od[d][i];

  // zero-fill strictly-upper attn region (reference exp underflows to exactly 0 there)
  const int z0 = qb + 16;
  if (z0 < T_) {
    float4 zz = {0.f, 0.f, 0.f, 0.f};
    const int n4 = (T_ - z0) >> 2;
    for (int ri = 0; ri < 16; ++ri) {
      float4* p = reinterpret_cast<float4*>(attn + (size_t)(qb + ri) * T_ + z0);
      for (int j = lane; j < n4; j += 32) p[j] = zz;
    }
  }
}

// ---------------- Kernel 3: mean over heads, then x Wo -------------------------------------
__global__ __launch_bounds__(32) void out_kernel(
    const float* __restrict__ heads, const float* __restrict__ Wo,
    float* __restrict__ outp) {
  const int tb = blockIdx.x * 16;      // global row in [0, B*T)
  const int lane = threadIdx.x;
  const int r = lane >> 4, ln = lane & 15;
  __shared__ unsigned short Mlds[16 * 64];

  for (int f = lane; f < 16 * 64; f += 32) {
    const int row16 = f >> 6, d = f & 63;
    const int gr = tb + row16;
    const int b = gr / T_, t = gr % T_;
    float ssum = 0.f;
#pragma unroll
    for (int hh = 0; hh < H_; ++hh)
      ssum += heads[((size_t)(b * H_ + hh) * T_ + t) * DK_ + d];
    Mlds[f] = f32_to_bf16(ssum * 0.125f);   // mean over 8 heads
  }
  __syncthreads();

  BfFrag A0, A1;
#pragma unroll
  for (int e = 0; e < 16; ++e) {
    A0.s[e] = Mlds[ln * 64 + a_k(e, r)];
    A1.s[e] = Mlds[ln * 64 + 32 + a_k(e, r)];
  }
  v8f z = {0.f, 0.f, 0.f, 0.f, 0.f, 0.f, 0.f, 0.f};
  for (int nt = 0; nt < DM_ / 16; ++nt) {
    v8f c = z;
    BfFrag Bf;
#pragma unroll
    for (int e = 0; e < 16; ++e)
      Bf.s[e] = f32_to_bf16(Wo[(size_t)(16 * r + e) * DM_ + nt * 16 + ln]);
    c = wmma_bf16(A0, Bf, c);
#pragma unroll
    for (int e = 0; e < 16; ++e)
      Bf.s[e] = f32_to_bf16(Wo[(size_t)(32 + 16 * r + e) * DM_ + nt * 16 + ln]);
    c = wmma_bf16(A1, Bf, c);
#pragma unroll
    for (int i = 0; i < 8; ++i)
      outp[(size_t)(tb + i + 8 * r) * DM_ + nt * 16 + ln] = c[i];
  }
}

extern "C" void kernel_launch(void* const* d_in, const int* in_sizes, int n_in,
                              void* d_out, int out_size, void* d_ws, size_t ws_size,
                              hipStream_t stream) {
  (void)in_sizes; (void)n_in; (void)out_size; (void)ws_size;
  const float* q  = (const float*)d_in[0];
  const float* k  = (const float*)d_in[1];
  const float* v  = (const float*)d_in[2];
  // d_in[3] = causal_mask (unused: causality applied analytically, matches mask exactly)
  const float* Wq = (const float*)d_in[4];
  const float* Wk = (const float*)d_in[5];
  const float* Wv = (const float*)d_in[6];
  const float* Wo = (const float*)d_in[7];

  // workspace layout (bf16 projections + fp32 heads), ~17.3 MB
  unsigned short* qs = (unsigned short*)d_ws;
  unsigned short* ks = qs + (size_t)B_ * H_ * T_ * DK_;
  unsigned short* vs = ks + (size_t)B_ * H_ * T_ * DK_;
  float* heads = reinterpret_cast<float*>(vs + (size_t)B_ * T_ * DK_);

  float* outp = (float*)d_out;                       // [B,T,512]
  float* attn = outp + (size_t)B_ * T_ * DM_;        // [H,B,T,T]

  proj_kernel<<<dim3(T_ / 16, 2 * H_ + 1, B_), 32, 0, stream>>>(q, k, v, Wq, Wk, Wv,
                                                                qs, ks, vs);
  attn_kernel<<<dim3(T_ / 16, H_, B_), 32, 0, stream>>>(qs, ks, vs, heads, attn);
  out_kernel<<<dim3((B_ * T_) / 16), 32, 0, stream>>>(heads, Wo, outp);
}
